// STGATBlock_20263655703138
// MI455X (gfx1250) — matrix operand order repaired and checked by reading, not compiled, not run on hardware
//
#include <hip/hip_runtime.h>
#include <math.h>
#include <stdint.h>

// ---------------------------------------------------------------------------
// Model constants (match reference)
// ---------------------------------------------------------------------------
constexpr int Dm   = 512;    // model dim
constexpr int Hh   = 8;      // heads
constexpr int Sq   = 64;     // src seq len
constexpr int Tq   = 32;     // tgt seq len
constexpr int NN_  = 512;    // nodes (batch)
constexpr int INC  = 32;     // in channels
constexpr int FEAT_= 64;
constexpr int NL_  = 4;
constexpr int FF_  = 2048;
constexpr int EE   = 16384;  // edges
constexpr float NEG_SLOPE = 0.2f;

typedef __attribute__((ext_vector_type(16))) _Float16 v16h;
typedef __attribute__((ext_vector_type(8)))  _Float16 v8h;
typedef __attribute__((ext_vector_type(8)))  float    v8f;

// ---------------------------------------------------------------------------
// Weight convert + transpose: wt[n*K + k] = (f16) w[k*ldw + n]
// (run once per weight per launch; makes the GEMM B operand f16 so B tiles
//  can be staged with gfx1250 async LDS copies, and halves B traffic)
// ---------------------------------------------------------------------------
__global__ void wcvt_kernel(const float* __restrict__ w, _Float16* __restrict__ wt,
                            int K, int N, int ldw)
{
    const int gid = blockIdx.x * blockDim.x + threadIdx.x;
    if (gid >= N * K) return;
    const int n = gid / K, k = gid % K;
    wt[gid] = (_Float16)w[(size_t)k * ldw + n];
}

// ---------------------------------------------------------------------------
// WMMA GEMM: C[M,N] = act(A[M,K] @ Bt^T[K,N] + bias[N])
//  - A: fp32 (M,K) row-major; converted to f16 through a register-staged,
//    software-pipelined path into LDS.
//  - Bt: f16 (N,K) row-major (pre-transposed weights); tiles staged with
//    GLOBAL_LOAD_ASYNC_TO_LDS_B128 into a double-buffered LDS tile,
//    synchronized with s_wait_asynccnt (per-wave) + workgroup barrier.
// Block tile 128x64x32, 128 threads = 4 waves; each wave: 2 A-frags x
// 4 B-frags = 8 v_wmma_f32_16x16x32_f16 per K-step.
// Requires M%128==0, N%64==0, K%32==0 (true for every GEMM here).
// act: 0 = none, 1 = relu
// ---------------------------------------------------------------------------
__global__ __launch_bounds__(128)
void gemm_wmma_kernel(const float* __restrict__ A, const _Float16* __restrict__ Bt,
                      const float* __restrict__ bias, float* __restrict__ C,
                      int M, int N, int K, int lda, int ldb, int ldc, int act)
{
    __shared__ _Float16 As[128][40];      // [m][k], 80B rows keep 16B alignment
    __shared__ _Float16 Bs[2][64][40];    // double-buffered [n][k] tiles

    const int tid  = threadIdx.x;
    const int lane = tid & 31;
    const int wv   = tid >> 5;            // wave id 0..3 -> rows wv*32..wv*32+31
    const int bm   = blockIdx.y * 128;
    const int bn   = blockIdx.x * 64;

    const int m16  = lane & 15;
    const int hi   = lane >> 4;           // 0/1 -> which K half this lane holds

    v8f acc[2][4] = {};                   // 2 row tiles x 4 col tiles (16x16 f32)

    // A loader geometry: 16 rows/pass, 8 passes, 8 float4 per row
    const int arow = tid >> 3;
    const int ac4  = tid & 7;
    float4 aReg[8];

    auto load_A = [&](int k0) {
        #pragma unroll
        for (int p = 0; p < 8; ++p)
            aReg[p] = *(const float4*)(A + (size_t)(bm + arow + p * 16) * lda + k0 + ac4 * 4);
    };
    auto store_A = [&]() {
        #pragma unroll
        for (int p = 0; p < 8; ++p) {
            _Float16* d = &As[arow + p * 16][ac4 * 4];
            d[0] = (_Float16)aReg[p].x; d[1] = (_Float16)aReg[p].y;
            d[2] = (_Float16)aReg[p].z; d[3] = (_Float16)aReg[p].w;
        }
    };

    // B tile: 64 rows x 32 halves = 256 x 16B segments; 2 per thread.
    // Async copy straight from global f16 into LDS (no VGPR round-trip).
    auto async_load_B = [&](int k0, int buf) {
        #pragma unroll
        for (int s = 0; s < 2; ++s) {
            const int seg = tid + s * 128;       // 0..255
            const int n   = seg >> 2;
            const int ch  = seg & 3;             // 16B chunk within row
            const unsigned long long gsrc =
                (unsigned long long)(uintptr_t)(Bt + (size_t)(bn + n) * ldb + k0 + ch * 8);
            const unsigned ldsoff =
                (unsigned)(uintptr_t)(const void*)&Bs[buf][n][ch * 8];
            asm volatile("global_load_async_to_lds_b128 %0, %1, off"
                         :: "v"(ldsoff), "v"(gsrc) : "memory");
        }
    };

    async_load_B(0, 0);
    load_A(0);

    int buf = 0;
    for (int k0 = 0; k0 < K; k0 += 32) {
        store_A();                                        // A regs -> LDS (f16)
        asm volatile("s_wait_asynccnt 0x0" ::: "memory"); // our B segments landed
        __syncthreads();                                  // whole tile visible

        if (k0 + 32 < K) {                                // overlap next tile with WMMAs
            async_load_B(k0 + 32, buf ^ 1);
            load_A(k0 + 32);
        }

        // A fragments (16x32 f16: lane<16 holds K{0..7,16..23}, lane>=16 +8)
        const v8h a00 = *(const v8h*)&As[wv * 32 + m16][hi * 8];
        const v8h a01 = *(const v8h*)&As[wv * 32 + m16][16 + hi * 8];
        const v16h af0 = __builtin_shufflevector(a00, a01, 0,1,2,3,4,5,6,7,8,9,10,11,12,13,14,15);
        const v8h a10 = *(const v8h*)&As[wv * 32 + 16 + m16][hi * 8];
        const v8h a11 = *(const v8h*)&As[wv * 32 + 16 + m16][16 + hi * 8];
        const v16h af1 = __builtin_shufflevector(a10, a11, 0,1,2,3,4,5,6,7,8,9,10,11,12,13,14,15);

        #pragma unroll
        for (int ct = 0; ct < 4; ++ct) {
            // B fragment (32x16: lane<16 K{0..7,16..23}, lane>=16 +8; col = lane&15)
            const v8h b0 = *(const v8h*)&Bs[buf][ct * 16 + m16][hi * 8];
            const v8h b1 = *(const v8h*)&Bs[buf][ct * 16 + m16][16 + hi * 8];
            const v16h bf = __builtin_shufflevector(b0, b1, 0,1,2,3,4,5,6,7,8,9,10,11,12,13,14,15);
            acc[0][ct] = __builtin_amdgcn_wmma_f32_16x16x32_f16(
                false, af0, false, bf, (short)0, acc[0][ct], false, false);
            acc[1][ct] = __builtin_amdgcn_wmma_f32_16x16x32_f16(
                false, af1, false, bf, (short)0, acc[1][ct], false, false);
        }
        __syncthreads();
        buf ^= 1;
    }

    // epilogue: C/D layout — VGPR r: M = r + 8*hi, N = lane&15
    #pragma unroll
    for (int rf = 0; rf < 2; ++rf) {
        #pragma unroll
        for (int ct = 0; ct < 4; ++ct) {
            #pragma unroll
            for (int r = 0; r < 8; ++r) {
                const int gr = bm + wv * 32 + rf * 16 + r + 8 * hi;
                const int gc = bn + ct * 16 + m16;
                float v = acc[rf][ct][r];
                if (bias) v += bias[gc];
                if (act == 1) v = fmaxf(v, 0.f);
                C[(size_t)gr * ldc + gc] = v;
            }
        }
    }
}

// ---------------------------------------------------------------------------
// Sinusoidal positional encoding add: x[l,n,d] += pe(l,d)
// ---------------------------------------------------------------------------
__global__ void posenc_kernel(float* __restrict__ x, int L)
{
    const size_t gid = (size_t)blockIdx.x * blockDim.x + threadIdx.x;
    if (gid >= (size_t)L * NN_ * Dm) return;
    const int d = (int)(gid % Dm);
    const int l = (int)(gid / ((size_t)Dm * NN_));
    const int d2 = d & ~1;
    const float freq = __expf(-9.210340371976184f * (float)d2 / (float)Dm); // ln(10000)
    const float ang  = (float)l * freq;
    x[gid] += (d & 1) ? __cosf(ang) : __sinf(ang);
}

// ---------------------------------------------------------------------------
// Fused residual + LayerNorm over D=512: out = LN(x [+ res]) * g + b
// ---------------------------------------------------------------------------
__global__ __launch_bounds__(128)
void ln_kernel(const float* __restrict__ x, const float* __restrict__ res,
               const float* __restrict__ g, const float* __restrict__ b,
               float* __restrict__ out)
{
    __shared__ float s1[128], s2[128];
    const size_t row = blockIdx.x;
    const int tid = threadIdx.x;
    float v[4], a1 = 0.f, a2 = 0.f;
    #pragma unroll
    for (int j = 0; j < 4; ++j) {
        const int c = j * 128 + tid;
        float t = x[row * Dm + c];
        if (res) t += res[row * Dm + c];
        v[j] = t; a1 += t; a2 += t * t;
    }
    s1[tid] = a1; s2[tid] = a2;
    __syncthreads();
    for (int st = 64; st > 0; st >>= 1) {
        if (tid < st) { s1[tid] += s1[tid + st]; s2[tid] += s2[tid + st]; }
        __syncthreads();
    }
    const float mean = s1[0] * (1.f / Dm);
    const float var  = s2[0] * (1.f / Dm) - mean * mean;
    const float rstd = rsqrtf(var + 1e-5f);
    #pragma unroll
    for (int j = 0; j < 4; ++j) {
        const int c = j * 128 + tid;
        out[row * Dm + c] = (v[j] - mean) * rstd * g[c] + b[c];
    }
}

// ---------------------------------------------------------------------------
// Attention: per (node n, head h) block; seq-first rows (l*NN + n).
// ---------------------------------------------------------------------------
template<int LQ, int LK, bool CAUSAL>
__global__ __launch_bounds__(64)
void attn_kernel(const float* __restrict__ Qm, int qld,
                 const float* __restrict__ Km, int kld,
                 const float* __restrict__ Vm, int vld,
                 float* __restrict__ Om, int ldo, float scale)
{
    __shared__ float qs[LQ * 64], ks[LK * 64], vs[LK * 64];
    const int n = blockIdx.x, h = blockIdx.y, tid = threadIdx.x;

    for (int idx = tid; idx < LQ * 64; idx += 64) {
        const int l = idx >> 6, d = idx & 63;
        qs[idx] = Qm[((size_t)l * NN_ + n) * qld + h * 64 + d];
    }
    for (int idx = tid; idx < LK * 64; idx += 64) {
        const int l = idx >> 6, d = idx & 63;
        ks[idx] = Km[((size_t)l * NN_ + n) * kld + h * 64 + d];
        vs[idx] = Vm[((size_t)l * NN_ + n) * vld + h * 64 + d];
    }
    __syncthreads();

    const int t = tid;
    if (t >= LQ) return;

    float s[LK];
    float mx = -1e30f;
    #pragma unroll
    for (int kk = 0; kk < LK; ++kk) {
        float a = 0.f;
        #pragma unroll 16
        for (int d = 0; d < 64; ++d) a += qs[t * 64 + d] * ks[kk * 64 + d];
        a *= scale;
        if (CAUSAL && kk > t) a = -1e30f;
        s[kk] = a;
        mx = fmaxf(mx, a);
    }
    float sum = 0.f;
    #pragma unroll
    for (int kk = 0; kk < LK; ++kk) { s[kk] = __expf(s[kk] - mx); sum += s[kk]; }
    const float inv = 1.f / sum;
    for (int d = 0; d < 64; ++d) {
        float o = 0.f;
        #pragma unroll
        for (int kk = 0; kk < LK; ++kk) o += s[kk] * vs[kk * 64 + d];
        Om[((size_t)t * NN_ + n) * ldo + h * 64 + d] = o * inv;
    }
}

// ---------------------------------------------------------------------------
// GATv2 kernels. Both layers use H=8, C=32 (H*C = 256).
// ---------------------------------------------------------------------------
__device__ __forceinline__ void atomicMaxF(float* addr, float val)
{
    if (__float_as_int(val) >= 0) atomicMax((int*)addr, __float_as_int(val));
    else                          atomicMin((unsigned int*)addr, __float_as_uint(val));
}

constexpr int TEH = Tq * EE * Hh;   // 4,194,304

__global__ void gat_edge_kernel(const float* __restrict__ xl, const float* __restrict__ xr,
                                const float* __restrict__ att, const int* __restrict__ esrc,
                                const int* __restrict__ edst, float* __restrict__ esc,
                                float* __restrict__ mbuf)
{
    const int gid = blockIdx.x * blockDim.x + threadIdx.x;
    if (gid >= TEH) return;
    const int h = gid % Hh;
    const int e = (gid / Hh) % EE;
    const int t = gid / (Hh * EE);
    const int src = esrc[e], dst = edst[e];
    const float* pl = xl + ((size_t)t * NN_ + src) * 256 + h * 32;
    const float* pr = xr + ((size_t)t * NN_ + dst) * 256 + h * 32;
    const float* pa = att + h * 32;
    float s = 0.f;
    #pragma unroll
    for (int c = 0; c < 32; ++c) {
        float z = pl[c] + pr[c];
        z = (z > 0.f) ? z : NEG_SLOPE * z;     // leaky_relu
        s += z * pa[c];
    }
    esc[gid] = s;
    atomicMaxF(mbuf + ((size_t)t * NN_ + dst) * Hh + h, s);
}

__global__ void gat_exp_kernel(float* __restrict__ esc, const float* __restrict__ mbuf,
                               float* __restrict__ den, const int* __restrict__ edst)
{
    const int gid = blockIdx.x * blockDim.x + threadIdx.x;
    if (gid >= TEH) return;
    const int h = gid % Hh;
    const int e = (gid / Hh) % EE;
    const int t = gid / (Hh * EE);
    const int dst = edst[e];
    const float ex = __expf(esc[gid] - mbuf[((size_t)t * NN_ + dst) * Hh + h]);
    esc[gid] = ex;
    atomicAdd(den + ((size_t)t * NN_ + dst) * Hh + h, ex);
}

__global__ void gat_scatter_kernel(const float* __restrict__ esc, const float* __restrict__ den,
                                   const float* __restrict__ xl, const int* __restrict__ esrc,
                                   const int* __restrict__ edst, float* __restrict__ out)
{
    const int gid = blockIdx.x * blockDim.x + threadIdx.x;
    if (gid >= TEH) return;
    const int h = gid % Hh;
    const int e = (gid / Hh) % EE;
    const int t = gid / (Hh * EE);
    const int src = esrc[e], dst = edst[e];
    const float alpha = esc[gid] / den[((size_t)t * NN_ + dst) * Hh + h];
    const float* pl = xl + ((size_t)t * NN_ + src) * 256 + h * 32;
    float* po = out + ((size_t)t * NN_ + dst) * 256 + h * 32;
    #pragma unroll
    for (int c = 0; c < 32; ++c) atomicAdd(po + c, alpha * pl[c]);
}

// ---------------------------------------------------------------------------
// Small elementwise helpers
// ---------------------------------------------------------------------------
__global__ void fill_kernel(float* __restrict__ p, float v, size_t n)
{
    const size_t gid = (size_t)blockIdx.x * blockDim.x + threadIdx.x;
    if (gid < n) p[gid] = v;
}

__global__ void bias_add_kernel(float* __restrict__ p, const float* __restrict__ bias, size_t n)
{
    const size_t gid = (size_t)blockIdx.x * blockDim.x + threadIdx.x;
    if (gid < n) p[gid] += bias[gid % 256];
}

__global__ void gat_out_act_kernel(float* __restrict__ out, const float* __restrict__ bias, size_t n)
{
    const size_t gid = (size_t)blockIdx.x * blockDim.x + threadIdx.x;
    if (gid >= n) return;
    float v = out[gid] + bias[gid % 256];
    v = (v > 0.f) ? v : (__expf(v) - 1.f);   // elu
    out[gid] = fmaxf(v, 0.f);                // relu
}

// ---------------------------------------------------------------------------
// Host orchestration
// ---------------------------------------------------------------------------
extern "C" void kernel_launch(void* const* d_in, const int* in_sizes, int n_in,
                              void* d_out, int out_size, void* d_ws, size_t ws_size,
                              hipStream_t stream)
{
    (void)in_sizes; (void)n_in; (void)out_size; (void)ws_size;

    const float* src        = (const float*)d_in[0];
    const float* tgt        = (const float*)d_in[1];
    const float* w_enc_in   = (const float*)d_in[2];
    const float* b_enc_in   = (const float*)d_in[3];
    const float* w_dec_in   = (const float*)d_in[4];
    const float* b_dec_in   = (const float*)d_in[5];
    const float* w_map      = (const float*)d_in[6];
    const float* b_map      = (const float*)d_in[7];
    const float* enc_qkv_w  = (const float*)d_in[8];
    const float* enc_qkv_b  = (const float*)d_in[9];
    const float* enc_out_w  = (const float*)d_in[10];
    const float* enc_out_b  = (const float*)d_in[11];
    const float* enc_ff1_w  = (const float*)d_in[12];
    const float* enc_ff1_b  = (const float*)d_in[13];
    const float* enc_ff2_w  = (const float*)d_in[14];
    const float* enc_ff2_b  = (const float*)d_in[15];
    const float* enc_ln1_g  = (const float*)d_in[16];
    const float* enc_ln1_b  = (const float*)d_in[17];
    const float* enc_ln2_g  = (const float*)d_in[18];
    const float* enc_ln2_b  = (const float*)d_in[19];
    const float* enc_fn_g   = (const float*)d_in[20];
    const float* enc_fn_b   = (const float*)d_in[21];
    const float* dsa_qkv_w  = (const float*)d_in[22];
    const float* dsa_qkv_b  = (const float*)d_in[23];
    const float* dsa_out_w  = (const float*)d_in[24];
    const float* dsa_out_b  = (const float*)d_in[25];
    const float* dca_qkv_w  = (const float*)d_in[26];
    const float* dca_qkv_b  = (const float*)d_in[27];
    const float* dca_out_w  = (const float*)d_in[28];
    const float* dca_out_b  = (const float*)d_in[29];
    const float* dff1_w     = (const float*)d_in[30];
    const float* dff1_b     = (const float*)d_in[31];
    const float* dff2_w     = (const float*)d_in[32];
    const float* dff2_b     = (const float*)d_in[33];
    const float* dln1_g     = (const float*)d_in[34];
    const float* dln1_b     = (const float*)d_in[35];
    const float* dln2_g     = (const float*)d_in[36];
    const float* dln2_b     = (const float*)d_in[37];
    const float* dln3_g     = (const float*)d_in[38];
    const float* dln3_b     = (const float*)d_in[39];
    const float* dfn_g      = (const float*)d_in[40];
    const float* dfn_b      = (const float*)d_in[41];
    const float* g1_wl      = (const float*)d_in[42];
    const float* g1_wr      = (const float*)d_in[43];
    const float* g1_bl      = (const float*)d_in[44];
    const float* g1_br      = (const float*)d_in[45];
    const float* g1_att     = (const float*)d_in[46];
    const float* g1_bias    = (const float*)d_in[47];
    const float* g2_wl      = (const float*)d_in[48];
    const float* g2_wr      = (const float*)d_in[49];
    const float* g2_bl      = (const float*)d_in[50];
    const float* g2_br      = (const float*)d_in[51];
    const float* g2_att     = (const float*)d_in[52];
    const float* g2_bias    = (const float*)d_in[53];
    const int*   edge_index = (const int*)d_in[54];
    const int*   esrc = edge_index;
    const int*   edst = edge_index + EE;

    const int ME = Sq * NN_;   // 32768 encoder rows
    const int MD = Tq * NN_;   // 16384 decoder rows

    // ---- workspace layout (floats) ----
    float* ws = (float*)d_ws;
    size_t off = 0;
    auto take = [&](size_t n) { float* p = ws + off; off += n; return p; };
    float* X    = take((size_t)ME * Dm);        // encoder stream
    float* MEMB = take((size_t)ME * Dm);        // encoder memory
    float* Y    = take((size_t)MD * Dm);        // decoder stream
    float* QKV  = take((size_t)ME * FF_);       // shared QKV (ld 1536) / FFN (ld 2048) buffer
    float* AO   = take((size_t)ME * Dm);        // attention output
    float* TMP  = take((size_t)ME * Dm);        // projection temp for residual
    float* MAP  = take((size_t)MD * FEAT_);     // decoder -> FEAT
    float* XL   = take((size_t)MD * 256);
    float* XR   = take((size_t)MD * 256);
    float* H1   = take((size_t)MD * 256);
    float* ESC  = take((size_t)TEH);
    float* MB   = take((size_t)Tq * NN_ * Hh);
    float* DEN  = take((size_t)Tq * NN_ * Hh);
    _Float16* WT = (_Float16*)take(524288);     // transposed f16 weights (max 1M halves)
    float* OUTF = (float*)d_out;                // (T, NN, 256)

    const float scale = 0.125f;                 // 1/sqrt(64)
    auto cdiv = [](size_t a, size_t b) { return (unsigned)((a + b - 1) / b); };

    // convert weight (K,N,row-stride ldw) -> WT (N,K) f16
    auto conv = [&](const float* w, int K, int N, int ldw) {
        wcvt_kernel<<<cdiv((size_t)N * K, 256), 256, 0, stream>>>(w, WT, K, N, ldw);
    };
    // GEMM with pre-transposed f16 B (a row offset into Bt selects output columns)
    auto gemm = [&](const float* A, const _Float16* Bt, const float* bias, float* C,
                    int M, int N, int K, int lda, int ldc, int act) {
        dim3 g(N / 64, M / 128), b(128);
        gemm_wmma_kernel<<<g, b, 0, stream>>>(A, Bt, bias, C, M, N, K, lda, K, ldc, act);
    };

    // =========================== encoder ===========================
    conv(w_enc_in, INC, Dm, Dm);
    gemm(src, WT, b_enc_in, X, ME, Dm, INC, INC, Dm, 0);
    posenc_kernel<<<cdiv((size_t)ME * Dm, 256), 256, 0, stream>>>(X, Sq);

    for (int i = 0; i < NL_; ++i) {
        conv(enc_qkv_w + (size_t)i * Dm * 3 * Dm, Dm, 3 * Dm, 3 * Dm);
        gemm(X, WT, enc_qkv_b + (size_t)i * 3 * Dm, QKV, ME, 3 * Dm, Dm, Dm, 3 * Dm, 0);
        attn_kernel<Sq, Sq, false><<<dim3(NN_, Hh), 64, 0, stream>>>(
            QKV, 3 * Dm, QKV + Dm, 3 * Dm, QKV + 2 * Dm, 3 * Dm, AO, Dm, scale);
        conv(enc_out_w + (size_t)i * Dm * Dm, Dm, Dm, Dm);
        gemm(AO, WT, enc_out_b + (size_t)i * Dm, TMP, ME, Dm, Dm, Dm, Dm, 0);
        ln_kernel<<<ME, 128, 0, stream>>>(X, TMP, enc_ln1_g + (size_t)i * Dm, enc_ln1_b + (size_t)i * Dm, X);
        conv(enc_ff1_w + (size_t)i * Dm * FF_, Dm, FF_, FF_);
        gemm(X, WT, enc_ff1_b + (size_t)i * FF_, QKV, ME, FF_, Dm, Dm, FF_, 1 /*relu*/);
        conv(enc_ff2_w + (size_t)i * FF_ * Dm, FF_, Dm, Dm);
        gemm(QKV, WT, enc_ff2_b + (size_t)i * Dm, TMP, ME, Dm, FF_, FF_, Dm, 0);
        ln_kernel<<<ME, 128, 0, stream>>>(X, TMP, enc_ln2_g + (size_t)i * Dm, enc_ln2_b + (size_t)i * Dm, X);
    }
    ln_kernel<<<ME, 128, 0, stream>>>(X, nullptr, enc_fn_g, enc_fn_b, MEMB);

    // =========================== decoder ===========================
    conv(w_dec_in, FEAT_, Dm, Dm);
    gemm(tgt, WT, b_dec_in, Y, MD, Dm, FEAT_, FEAT_, Dm, 0);
    posenc_kernel<<<cdiv((size_t)MD * Dm, 256), 256, 0, stream>>>(Y, Tq);

    for (int i = 0; i < NL_; ++i) {
        // causal self-attention
        conv(dsa_qkv_w + (size_t)i * Dm * 3 * Dm, Dm, 3 * Dm, 3 * Dm);
        gemm(Y, WT, dsa_qkv_b + (size_t)i * 3 * Dm, QKV, MD, 3 * Dm, Dm, Dm, 3 * Dm, 0);
        attn_kernel<Tq, Tq, true><<<dim3(NN_, Hh), 64, 0, stream>>>(
            QKV, 3 * Dm, QKV + Dm, 3 * Dm, QKV + 2 * Dm, 3 * Dm, AO, Dm, scale);
        conv(dsa_out_w + (size_t)i * Dm * Dm, Dm, Dm, Dm);
        gemm(AO, WT, dsa_out_b + (size_t)i * Dm, TMP, MD, Dm, Dm, Dm, Dm, 0);
        ln_kernel<<<MD, 128, 0, stream>>>(Y, TMP, dln1_g + (size_t)i * Dm, dln1_b + (size_t)i * Dm, Y);

        // cross-attention: Q from Y, K/V from encoder memory.
        // One conversion of the whole (512,1536) wqkv serves all three GEMMs:
        // q -> WT rows [0,512), k -> rows [512,1024), v -> rows [1024,1536).
        const float* bc = dca_qkv_b + (size_t)i * 3 * Dm;
        conv(dca_qkv_w + (size_t)i * Dm * 3 * Dm, Dm, 3 * Dm, 3 * Dm);
        gemm(Y,    WT,                      bc,          QKV,          MD, Dm, Dm, Dm, 3 * Dm, 0);
        gemm(MEMB, WT + (size_t)Dm * Dm,     bc + Dm,     QKV + Dm,     ME, Dm, Dm, Dm, 3 * Dm, 0);
        gemm(MEMB, WT + (size_t)2 * Dm * Dm, bc + 2 * Dm, QKV + 2 * Dm, ME, Dm, Dm, Dm, 3 * Dm, 0);
        attn_kernel<Tq, Sq, false><<<dim3(NN_, Hh), 64, 0, stream>>>(
            QKV, 3 * Dm, QKV + Dm, 3 * Dm, QKV + 2 * Dm, 3 * Dm, AO, Dm, scale);
        conv(dca_out_w + (size_t)i * Dm * Dm, Dm, Dm, Dm);
        gemm(AO, WT, dca_out_b + (size_t)i * Dm, TMP, MD, Dm, Dm, Dm, Dm, 0);
        ln_kernel<<<MD, 128, 0, stream>>>(Y, TMP, dln2_g + (size_t)i * Dm, dln2_b + (size_t)i * Dm, Y);

        // FFN
        conv(dff1_w + (size_t)i * Dm * FF_, Dm, FF_, FF_);
        gemm(Y, WT, dff1_b + (size_t)i * FF_, QKV, MD, FF_, Dm, Dm, FF_, 1);
        conv(dff2_w + (size_t)i * FF_ * Dm, FF_, Dm, Dm);
        gemm(QKV, WT, dff2_b + (size_t)i * Dm, TMP, MD, Dm, FF_, FF_, Dm, 0);
        ln_kernel<<<MD, 128, 0, stream>>>(Y, TMP, dln3_g + (size_t)i * Dm, dln3_b + (size_t)i * Dm, Y);
    }
    ln_kernel<<<MD, 128, 0, stream>>>(Y, nullptr, dfn_g, dfn_b, Y);
    conv(w_map, Dm, FEAT_, FEAT_);
    gemm(Y, WT, b_map, MAP, MD, FEAT_, Dm, Dm, FEAT_, 0);

    // =========================== GATv2 ===========================
    const size_t TNH = (size_t)Tq * NN_ * Hh;
    const unsigned EB = cdiv(TEH, 256);

    // layer 1: (T*NN, 64) -> (T*NN, 256)
    conv(g1_wl, FEAT_, 256, 256);
    gemm(MAP, WT, g1_bl, XL, MD, 256, FEAT_, FEAT_, 256, 0);
    conv(g1_wr, FEAT_, 256, 256);
    gemm(MAP, WT, g1_br, XR, MD, 256, FEAT_, FEAT_, 256, 0);
    fill_kernel<<<cdiv(TNH, 256), 256, 0, stream>>>(MB, -3.0e38f, TNH);
    fill_kernel<<<cdiv(TNH, 256), 256, 0, stream>>>(DEN, 0.f, TNH);
    fill_kernel<<<cdiv((size_t)MD * 256, 256), 256, 0, stream>>>(H1, 0.f, (size_t)MD * 256);
    gat_edge_kernel<<<EB, 256, 0, stream>>>(XL, XR, g1_att, esrc, edst, ESC, MB);
    gat_exp_kernel<<<EB, 256, 0, stream>>>(ESC, MB, DEN, edst);
    gat_scatter_kernel<<<EB, 256, 0, stream>>>(ESC, DEN, XL, esrc, edst, H1);
    bias_add_kernel<<<cdiv((size_t)MD * 256, 256), 256, 0, stream>>>(H1, g1_bias, (size_t)MD * 256);

    // layer 2: (T*NN, 256) -> (T*NN, 256), then elu + relu
    conv(g2_wl, 256, 256, 256);
    gemm(H1, WT, g2_bl, XL, MD, 256, 256, 256, 256, 0);
    conv(g2_wr, 256, 256, 256);
    gemm(H1, WT, g2_br, XR, MD, 256, 256, 256, 256, 0);
    fill_kernel<<<cdiv(TNH, 256), 256, 0, stream>>>(MB, -3.0e38f, TNH);
    fill_kernel<<<cdiv(TNH, 256), 256, 0, stream>>>(DEN, 0.f, TNH);
    fill_kernel<<<cdiv((size_t)MD * 256, 256), 256, 0, stream>>>(OUTF, 0.f, (size_t)MD * 256);
    gat_edge_kernel<<<EB, 256, 0, stream>>>(XL, XR, g2_att, esrc, edst, ESC, MB);
    gat_exp_kernel<<<EB, 256, 0, stream>>>(ESC, MB, DEN, edst);
    gat_scatter_kernel<<<EB, 256, 0, stream>>>(ESC, DEN, XL, esrc, edst, OUTF);
    gat_out_act_kernel<<<cdiv((size_t)MD * 256, 256), 256, 0, stream>>>(OUTF, g2_bias, (size_t)MD * 256);
}
